// SimpleDiffusionBlock_65335042506800
// MI455X (gfx1250) — matrix operand order repaired and checked by reading, not compiled, not run on hardware
//
#include <hip/hip_runtime.h>

typedef __bf16 bf16;
typedef __attribute__((ext_vector_type(16))) __bf16 bf16x16;
typedef __attribute__((ext_vector_type(8)))  __bf16 bf16x8;
typedef __attribute__((ext_vector_type(8)))  float  f32x8;
typedef __attribute__((ext_vector_type(4)))  float  f32x4;
typedef __attribute__((ext_vector_type(4)))  unsigned int u32x4;
typedef __attribute__((ext_vector_type(8)))  int i32x8;
typedef __attribute__((ext_vector_type(4)))  int i32x4;

#define NTOK 1536
#define DIM  1024
#define NH   16
#define HK   64
#define NPTS 8
#define FD   5376       // 42 * 128
#define D2   2048
#define KHAT 96         // 64 + 24 + 8 pad

// ---------------------------------------------------------------------------
// WMMA helpers (CDNA5 / gfx1250, wave32)
// ---------------------------------------------------------------------------
__device__ __forceinline__ f32x8 wmma_bf16(bf16x16 a, bf16x16 b, f32x8 c) {
  return __builtin_amdgcn_wmma_f32_16x16x32_bf16(
      false, a, false, b, (short)0, c, false, false);
}

// A fragment (16x32 bf16): lane holds row m=lane&15; K-chunks at
// c0+half*8 and c0+16+half*8 (16B-aligned ds_load_b128).
__device__ __forceinline__ bf16x16 load_a_frag(const bf16* base, int stride,
                                               int c0, int lane) {
  const int m = lane & 15, half = (lane >> 4) & 1;
  const bf16* p = base + m * stride + c0 + half * 8;
  bf16x8 lo = *(const bf16x8*)(p);
  bf16x8 hi = *(const bf16x8*)(p + 16);
  bf16x16 r;
#pragma unroll
  for (int i = 0; i < 8; ++i) { r[i] = lo[i]; r[i + 8] = hi[i]; }
  return r;
}

// B fragment (32x16 bf16) from a tile stored [n][k] (k contiguous):
// lane holds column n=lane&15; K-chunks at c0+half*8 and c0+16+half*8.
__device__ __forceinline__ bf16x16 load_b_frag(const bf16* base, int stride,
                                               int c0, int lane) {
  const int n = lane & 15, half = (lane >> 4) & 1;
  const bf16* p = base + n * stride + c0 + half * 8;
  bf16x8 lo = *(const bf16x8*)(p);
  bf16x8 hi = *(const bf16x8*)(p + 16);
  bf16x16 r;
#pragma unroll
  for (int i = 0; i < 8; ++i) { r[i] = lo[i]; r[i + 8] = hi[i]; }
  return r;
}

// ---------------------------------------------------------------------------
// Tensor Data Mover: DMA a [128 rows x 32 elems] bf16 tile into LDS with
// 16B padding after each 64B row -> LDS row stride 80B ([r][40] bf16 layout).
// rows_avail = remaining rows for OOB zero-fill; stride in elements.
// ---------------------------------------------------------------------------
#define HAVE_TDM __has_builtin(__builtin_amdgcn_tensor_load_to_lds)

#if HAVE_TDM
__device__ __forceinline__ void tdm_load_tile(unsigned lds_off, const bf16* g,
                                              int rows_avail, int row_len,
                                              int row_stride) {
  unsigned long long ga = (unsigned long long)(size_t)g;
  u32x4 g0;
  g0[0] = 1u;                                       // count=1, user descriptor
  g0[1] = lds_off;                                  // LDS byte address
  g0[2] = (unsigned)(ga & 0xFFFFFFFFu);             // global_addr[31:0]
  g0[3] = (unsigned)((ga >> 32) & 0x1FFFFFFu) | (2u << 30);  // [56:32] | type=2
  i32x8 g1;
  g1[0] = (1 << 16)                                 // data_size = 2 bytes
        | (1 << 20)                                 // pad_enable
        | (3 << 22)                                 // pad_interval: 16 DWORDs
        | (3 << 25);                                // pad_amount: 4 DWORDs
  g1[1] = (row_len & 0xFFFF) << 16;                 // tensor_dim0[15:0]
  g1[2] = ((row_len >> 16) & 0xFFFF)                // tensor_dim0[31:16]
        | ((rows_avail & 0xFFFF) << 16);            // tensor_dim1[15:0]
  g1[3] = ((rows_avail >> 16) & 0xFFFF)             // tensor_dim1[31:16]
        | (32 << 16);                               // tile_dim0 = 32 elems
  g1[4] = 128;                                      // tile_dim1 = 128 rows
  g1[5] = row_stride;                               // tensor_dim0_stride lo
  g1[6] = 0;
  g1[7] = 0;
  i32x4 z4 = {0, 0, 0, 0};
  i32x8 z8 = {0, 0, 0, 0, 0, 0, 0, 0};
  __builtin_amdgcn_tensor_load_to_lds(g0, g1, z4, z4, z8, 0);
}
#endif

__device__ __forceinline__ unsigned lds_addr_of(const void* p) {
  return (unsigned)(unsigned long long)(size_t)p;
}

// ---------------------------------------------------------------------------
// Tiled GEMM:  C[M,N] = act(A[M,K] . B[K,N] + bias) + res
// A  : bf16 row-major [M][K]
// Bt : bf16 pre-transposed [N][K]  (== the LDS-friendly [n][k] layout)
// Cout: f32 (outbf=0) or bf16 (outbf=1).  K % 32 == 0.
// Tiles stream in via TDM (double buffered), issued by wave 0.
// ---------------------------------------------------------------------------
#define GTM 128
#define GTN 128
#define GTK 32

__global__ __launch_bounds__(256) void gemm_kernel(
    const bf16* __restrict__ A, const bf16* __restrict__ Bt,
    const float* __restrict__ bias, const float* __restrict__ res,
    void* __restrict__ Cout, int M, int N, int K, int act, int outbf) {
  __shared__ bf16 As[2][GTM][40];
  __shared__ bf16 Bs[2][GTN][40];
  const int tid = threadIdx.x;
  const int lane = tid & 31, wid = tid >> 5;
  const int m0 = blockIdx.y * GTM, n0 = blockIdx.x * GTN;
  const int wm = (wid >> 1) * 32, wn = (wid & 1) * 64;
  const int hn = lane & 15, half = (lane >> 4) & 1;
  const int nk = K / GTK;

  f32x8 acc[2][4];
  const f32x8 zero8 = {0.f, 0.f, 0.f, 0.f, 0.f, 0.f, 0.f, 0.f};
#pragma unroll
  for (int i = 0; i < 2; ++i)
#pragma unroll
    for (int j = 0; j < 4; ++j) acc[i][j] = zero8;

#if HAVE_TDM
  if (wid == 0) {
    tdm_load_tile(lds_addr_of(&As[0][0][0]), A + (size_t)m0 * K, M - m0, K, K);
    tdm_load_tile(lds_addr_of(&Bs[0][0][0]), Bt + (size_t)n0 * K, N - n0, K, K);
  }
  for (int kt = 0; kt < nk; ++kt) {
    const int buf = kt & 1;
    if (wid == 0) {
      if (kt + 1 < nk) {
        const int ko = (kt + 1) * GTK;
        tdm_load_tile(lds_addr_of(&As[buf ^ 1][0][0]),
                      A + (size_t)m0 * K + ko, M - m0, K - ko, K);
        tdm_load_tile(lds_addr_of(&Bs[buf ^ 1][0][0]),
                      Bt + (size_t)n0 * K + ko, N - n0, K - ko, K);
        __builtin_amdgcn_s_wait_tensorcnt(2);   // current tile complete
      } else {
        __builtin_amdgcn_s_wait_tensorcnt(0);
      }
    }
    __syncthreads();
    bf16x16 afr[2], bfr[4];
    afr[0] = load_a_frag(&As[buf][wm][0],      40, 0, lane);
    afr[1] = load_a_frag(&As[buf][wm + 16][0], 40, 0, lane);
#pragma unroll
    for (int j = 0; j < 4; ++j)
      bfr[j] = load_b_frag(&Bs[buf][wn + 16 * j][0], 40, 0, lane);
#pragma unroll
    for (int i = 0; i < 2; ++i)
#pragma unroll
      for (int j = 0; j < 4; ++j)
        acc[i][j] = wmma_bf16(afr[i], bfr[j], acc[i][j]);
    __syncthreads();
  }
#else
  // Fallback: cooperative VGPR staging (no TDM builtin on this toolchain).
  for (int kt = 0; kt < nk; ++kt) {
    const int ko = kt * GTK;
    __syncthreads();
#pragma unroll
    for (int it = 0; it < 2; ++it) {
      int slot = tid + 256 * it;                 // 512 bf16x8 slots per tile
      int r = slot >> 2, c8 = (slot & 3) * 8;
      int gm = m0 + r, gn = n0 + r;
      bf16x8 av = {};
      if (gm < M) av = *(const bf16x8*)(A + (size_t)gm * K + ko + c8);
      *(bf16x8*)(&As[0][r][c8]) = av;
      bf16x8 bv = {};
      if (gn < N) bv = *(const bf16x8*)(Bt + (size_t)gn * K + ko + c8);
      *(bf16x8*)(&Bs[0][r][c8]) = bv;
    }
    __syncthreads();
    bf16x16 afr[2], bfr[4];
    afr[0] = load_a_frag(&As[0][wm][0],      40, 0, lane);
    afr[1] = load_a_frag(&As[0][wm + 16][0], 40, 0, lane);
#pragma unroll
    for (int j = 0; j < 4; ++j)
      bfr[j] = load_b_frag(&Bs[0][wn + 16 * j][0], 40, 0, lane);
#pragma unroll
    for (int i = 0; i < 2; ++i)
#pragma unroll
      for (int j = 0; j < 4; ++j)
        acc[i][j] = wmma_bf16(afr[i], bfr[j], acc[i][j]);
  }
#endif

  float* Cf = (float*)Cout;
  bf16* Cb = (bf16*)Cout;
#pragma unroll
  for (int i = 0; i < 2; ++i) {
#pragma unroll
    for (int j = 0; j < 4; ++j) {
#pragma unroll
      for (int v = 0; v < 8; ++v) {
        int gm = m0 + wm + i * 16 + v + 8 * half;
        int gn = n0 + wn + j * 16 + hn;
        if (gm < M && gn < N) {
          float val = acc[i][j][v];
          if (bias) val += bias[gn];
          if (act == 1) val = val / (1.f + __expf(-val));          // silu
          else if (act == 2) val = 1.f / (1.f + __expf(-val));     // sigmoid
          if (res) val += res[(size_t)gm * N + gn];
          if (outbf) Cb[(size_t)gm * N + gn] = (bf16)val;
          else       Cf[(size_t)gm * N + gn] = val;
        }
      }
    }
  }
}

// ---------------------------------------------------------------------------
// Fused flash-style IPA attention.
// ---------------------------------------------------------------------------
__global__ __launch_bounds__(256) void attn_kernel(
    const bf16* __restrict__ Ahat, const bf16* __restrict__ Bhat,
    const bf16* __restrict__ Vhat, const float* __restrict__ an,
    const float* __restrict__ bn, const float* __restrict__ pair_bias,
    const unsigned char* __restrict__ mask, float* __restrict__ attn_out,
    int N) {
  __shared__ bf16 Bs[32][KHAT];
  __shared__ bf16 Vt[KHAT][40];
  __shared__ bf16 Pw[8][16][40];
  const int tid = threadIdx.x, lane = tid & 31, wid = tid >> 5;
  const int h = blockIdx.y;
  const int i0 = blockIdx.x * 128 + wid * 16;
  const int hn = lane & 15, half = (lane >> 4) & 1;

  bf16x16 afr[3];
#pragma unroll
  for (int cs = 0; cs < 3; ++cs)
    afr[cs] = load_a_frag(Ahat + ((size_t)h * N + i0) * KHAT, KHAT, cs * 32, lane);

  float anv[8], mrow[8], lrow[8];
#pragma unroll
  for (int v = 0; v < 8; ++v) {
    anv[v] = an[h * N + i0 + v + 8 * half];
    mrow[v] = -1e30f;
    lrow[v] = 0.f;
  }
  const f32x8 zero8 = {0.f, 0.f, 0.f, 0.f, 0.f, 0.f, 0.f, 0.f};
  f32x8 o[6];
#pragma unroll
  for (int dt = 0; dt < 6; ++dt) o[dt] = zero8;

  for (int jt = 0; jt < N; jt += 32) {
    __syncthreads();
#pragma unroll
    for (int it = 0; it < 12; ++it) {
      int e = tid + 256 * it;
      int r = e / KHAT, c = e % KHAT;
      size_t g = ((size_t)h * N + jt + r) * KHAT + c;
      Bs[r][c] = Bhat[g];
      Vt[c][r] = Vhat[g];
    }
    __syncthreads();

    f32x8 s[2];
    s[0] = zero8; s[1] = zero8;
#pragma unroll
    for (int jj = 0; jj < 2; ++jj)
#pragma unroll
      for (int cs = 0; cs < 3; ++cs)
        s[jj] = wmma_bf16(afr[cs],
                          load_b_frag(&Bs[jj * 16][0], KHAT, cs * 32, lane),
                          s[jj]);

#pragma unroll
    for (int jj = 0; jj < 2; ++jj) {
      int j = jt + jj * 16 + hn;
      float bnj = bn[h * N + j];
      bool mk = mask[j] != 0;
#pragma unroll
      for (int v = 0; v < 8; ++v) {
        int i = i0 + v + 8 * half;
        float val = s[jj][v] + pair_bias[((size_t)h * N + i) * N + j]
                    - anv[v] - bnj;
        s[jj][v] = mk ? val : -1e9f;
      }
    }

    float rmax[8], rsum[8], mnew[8], scl[8];
#pragma unroll
    for (int v = 0; v < 8; ++v) rmax[v] = fmaxf(s[0][v], s[1][v]);
#pragma unroll
    for (int mm = 1; mm < 16; mm <<= 1)
#pragma unroll
      for (int v = 0; v < 8; ++v)
        rmax[v] = fmaxf(rmax[v], __shfl_xor(rmax[v], mm, 32));
#pragma unroll
    for (int v = 0; v < 8; ++v) {
      mnew[v] = fmaxf(mrow[v], rmax[v]);
      scl[v] = __expf(mrow[v] - mnew[v]);
      mrow[v] = mnew[v];
    }
#pragma unroll
    for (int jj = 0; jj < 2; ++jj)
#pragma unroll
      for (int v = 0; v < 8; ++v) s[jj][v] = __expf(s[jj][v] - mnew[v]);
#pragma unroll
    for (int v = 0; v < 8; ++v) rsum[v] = s[0][v] + s[1][v];
#pragma unroll
    for (int mm = 1; mm < 16; mm <<= 1)
#pragma unroll
      for (int v = 0; v < 8; ++v) rsum[v] += __shfl_xor(rsum[v], mm, 32);
#pragma unroll
    for (int v = 0; v < 8; ++v) lrow[v] = lrow[v] * scl[v] + rsum[v];
#pragma unroll
    for (int dt = 0; dt < 6; ++dt)
#pragma unroll
      for (int v = 0; v < 8; ++v) o[dt][v] *= scl[v];

#pragma unroll
    for (int jj = 0; jj < 2; ++jj)
#pragma unroll
      for (int v = 0; v < 8; ++v)
        Pw[wid][v + 8 * half][jj * 16 + hn] = (bf16)s[jj][v];
    bf16x16 pf = load_a_frag(&Pw[wid][0][0], 40, 0, lane);
#pragma unroll
    for (int dt = 0; dt < 6; ++dt)
      o[dt] = wmma_bf16(pf, load_b_frag(&Vt[dt * 16][0], 40, 0, lane), o[dt]);
  }

  float invl[8];
#pragma unroll
  for (int v = 0; v < 8; ++v) invl[v] = 1.f / lrow[v];
#pragma unroll
  for (int dt = 0; dt < 6; ++dt)
#pragma unroll
    for (int v = 0; v < 8; ++v)
      attn_out[((size_t)h * N + i0 + v + 8 * half) * KHAT + dt * 16 + hn] =
          o[dt][v] * invl[v];
}

// ---------------------------------------------------------------------------
// Weight transpose+convert:  B[K][N] f32  ->  Bt[N][K] bf16
// ---------------------------------------------------------------------------
__global__ void tconv_kernel(const float* __restrict__ B, bf16* __restrict__ Bt,
                             int K, int N) {
  __shared__ float tile[32][33];
  int kb = blockIdx.y * 32, nb = blockIdx.x * 32;
  int tx = threadIdx.x, ty = threadIdx.y;
  for (int i = ty; i < 32; i += 8) {
    int k = kb + i, n = nb + tx;
    tile[i][tx] = (k < K && n < N) ? B[(size_t)k * N + n] : 0.f;
  }
  __syncthreads();
  for (int i = ty; i < 32; i += 8) {
    int n = nb + i, k = kb + tx;
    if (n < N && k < K) Bt[(size_t)n * K + k] = (bf16)tile[tx][i];
  }
}

__global__ void cvt_kernel(const float* __restrict__ a, bf16* __restrict__ o,
                           int n) {
  int i = blockIdx.x * blockDim.x + threadIdx.x;
  if (i < n) o[i] = (bf16)a[i];
}

// ---------------------------------------------------------------------------
// Small elementwise / per-row kernels
// ---------------------------------------------------------------------------
__global__ void embed_kernel(const float* __restrict__ pos,
                             bf16* __restrict__ emb) {
  int idx = blockIdx.x * blockDim.x + threadIdx.x;
  if (idx >= NTOK * 42 * 64) return;
  int n = idx / (42 * 64);
  int rem = idx - n * (42 * 64);
  int ti = rem >> 6, f = rem & 63;
  const float l0 = -6.6438561897747f;           // log2(0.01)
  const float step = (8.0f - l0) / 63.0f;       // to log2(256)
  float fr = exp2f(l0 + (float)f * step);
  float r = pos[n * 42 + ti] * fr;
  bf16* o = emb + (size_t)n * FD + ti * 128 + f * 2;
  o[0] = (bf16)__sinf(r);
  o[1] = (bf16)__cosf(r);
}

__global__ void ln_mod_kernel(const float* __restrict__ x,
                              const float* __restrict__ s1,
                              const float* __restrict__ s2,
                              bf16* __restrict__ out, int D) {
  int row = blockIdx.x, tid = threadIdx.x;
  __shared__ float red[256];
  float s = 0.f, q = 0.f;
  for (int c = tid; c < D; c += 256) {
    float v = x[(size_t)row * D + c];
    s += v; q += v * v;
  }
  red[tid] = s; __syncthreads();
  for (int st = 128; st > 0; st >>= 1) {
    if (tid < st) red[tid] += red[tid + st];
    __syncthreads();
  }
  float mean = red[0] / (float)D;
  __syncthreads();
  red[tid] = q; __syncthreads();
  for (int st = 128; st > 0; st >>= 1) {
    if (tid < st) red[tid] += red[tid + st];
    __syncthreads();
  }
  float var = red[0] / (float)D - mean * mean;
  float rstd = rsqrtf(var + 1e-5f);
  for (int c = tid; c < D; c += 256) {
    size_t i = (size_t)row * D + c;
    out[i] = (bf16)(s1[i] * ((x[i] - mean) * rstd) + s2[i]);
  }
}

__global__ void frames_kernel(const float* __restrict__ pos,
                              float* __restrict__ R, float* __restrict__ t) {
  int n = blockIdx.x * blockDim.x + threadIdx.x;
  if (n >= NTOK) return;
  const float* p = pos + (size_t)n * 42;
  float nn[3] = {p[0], p[1], p[2]};
  float ca[3] = {p[3], p[4], p[5]};
  float cc[3] = {p[6], p[7], p[8]};
  float e1[3], v2[3], e2[3], e3[3];
  float d = 0.f;
  for (int a = 0; a < 3; ++a) { e1[a] = cc[a] - ca[a]; d += e1[a] * e1[a]; }
  float r1 = rsqrtf(d + 1e-8f);
  for (int a = 0; a < 3; ++a) e1[a] *= r1;
  float dp = 0.f;
  for (int a = 0; a < 3; ++a) { v2[a] = nn[a] - ca[a]; dp += e1[a] * v2[a]; }
  d = 0.f;
  for (int a = 0; a < 3; ++a) { e2[a] = v2[a] - dp * e1[a]; d += e2[a] * e2[a]; }
  float r2 = rsqrtf(d + 1e-8f);
  for (int a = 0; a < 3; ++a) e2[a] *= r2;
  e3[0] = e1[1] * e2[2] - e1[2] * e2[1];
  e3[1] = e1[2] * e2[0] - e1[0] * e2[2];
  e3[2] = e1[0] * e2[1] - e1[1] * e2[0];
  for (int a = 0; a < 3; ++a) {
    R[n * 9 + a * 3 + 0] = e1[a];
    R[n * 9 + a * 3 + 1] = e2[a];
    R[n * 9 + a * 3 + 2] = e3[a];
    t[n * 3 + a] = ca[a];
  }
}

__global__ void toglob_kernel(float* __restrict__ buf,
                              const float* __restrict__ R,
                              const float* __restrict__ t) {
  int idx = blockIdx.x * blockDim.x + threadIdx.x;
  if (idx >= NTOK * NH * NPTS) return;
  int n = idx >> 7, r = idx & 127;
  float* p = buf + (size_t)n * 384 + r * 3;
  float x0 = p[0], x1 = p[1], x2 = p[2];
  const float* Rm = R + n * 9;
  const float* tl = t + n * 3;
#pragma unroll
  for (int a = 0; a < 3; ++a)
    p[a] = Rm[a * 3 + 0] * x0 + Rm[a * 3 + 1] * x1 + Rm[a * 3 + 2] * x2 + tl[a];
}

__global__ void pack_qkv_kernel(const float* __restrict__ q,
                                const float* __restrict__ k,
                                const float* __restrict__ v,
                                const float* __restrict__ qg,
                                const float* __restrict__ kg,
                                const float* __restrict__ vg,
                                const float* __restrict__ gamma,
                                bf16* __restrict__ Ahat, bf16* __restrict__ Bhat,
                                bf16* __restrict__ Vhat, float* __restrict__ an,
                                float* __restrict__ bn) {
  int idx = blockIdx.x * blockDim.x + threadIdx.x;
  if (idx >= NTOK * NH) return;
  int n = idx >> 4, h = idx & 15;
  float g = 0.5f * (1.f / 6.f) * log1pf(__expf(gamma[h]));  // wc = 1/6
  float sc = sqrtf(2.f * g);
  const float rsK = 0.125f;  // 1/sqrt(64)
  bf16* Ar = Ahat + ((size_t)h * NTOK + n) * KHAT;
  bf16* Br = Bhat + ((size_t)h * NTOK + n) * KHAT;
  bf16* Vr = Vhat + ((size_t)h * NTOK + n) * KHAT;
  size_t qb = (size_t)n * (NH * HK) + h * HK;
  for (int e = 0; e < HK; ++e) {
    Ar[e] = (bf16)(q[qb + e] * rsK);
    Br[e] = (bf16)(k[qb + e]);
    Vr[e] = (bf16)(v[qb + e]);
  }
  size_t pb = (size_t)n * 384 + h * 24;
  float qs = 0.f, ks = 0.f;
  for (int e = 0; e < 24; ++e) {
    float qv = qg[pb + e], kv = kg[pb + e], vv = vg[pb + e];
    Ar[64 + e] = (bf16)(sc * qv);
    Br[64 + e] = (bf16)(sc * kv);
    Vr[64 + e] = (bf16)vv;
    qs += qv * qv; ks += kv * kv;
  }
  for (int e = 88; e < KHAT; ++e) {
    Ar[e] = (bf16)0.f; Br[e] = (bf16)0.f; Vr[e] = (bf16)0.f;
  }
  an[h * NTOK + n] = g * qs;
  bn[h * NTOK + n] = g * ks;
}

__global__ void pack_upd_kernel(const float* __restrict__ attn,
                                const float* __restrict__ R,
                                const float* __restrict__ t,
                                bf16* __restrict__ upd) {
  int idx = blockIdx.x * blockDim.x + threadIdx.x;
  if (idx >= NTOK * NH) return;
  int n = idx >> 4, h = idx & 15;
  const float* ar = attn + ((size_t)h * NTOK + n) * KHAT;
  bf16* ur = upd + (size_t)n * 1536;
  for (int e = 0; e < HK; ++e) ur[h * HK + e] = (bf16)ar[e];
  const float* Rm = R + n * 9;
  const float* tl = t + n * 3;
  for (int p = 0; p < NPTS; ++p) {
    float xg[3], opl[3];
    for (int a = 0; a < 3; ++a) xg[a] = ar[64 + p * 3 + a] - tl[a];
    float s2 = 0.f;
    for (int a = 0; a < 3; ++a) {
      opl[a] = Rm[0 * 3 + a] * xg[0] + Rm[1 * 3 + a] * xg[1] + Rm[2 * 3 + a] * xg[2];
      ur[1024 + h * 24 + p * 3 + a] = (bf16)opl[a];
      s2 += opl[a] * opl[a];
    }
    ur[1408 + h * 8 + p] = (bf16)sqrtf(s2 + 1e-8f);
  }
}

__global__ void mul_kernel(const bf16* __restrict__ a,
                           const bf16* __restrict__ b,
                           bf16* __restrict__ o, int n) {
  int i = blockIdx.x * blockDim.x + threadIdx.x;
  if (i < n) o[i] = (bf16)((float)a[i] * (float)b[i]);
}

// ---------------------------------------------------------------------------
// Host-side orchestration
// ---------------------------------------------------------------------------
extern "C" void kernel_launch(void* const* d_in, const int* in_sizes, int n_in,
                              void* d_out, int out_size, void* d_ws,
                              size_t ws_size, hipStream_t stream) {
  const float* in_local = (const float*)d_in[0];
  const float* pos      = (const float*)d_in[1];
  const float* cond     = (const float*)d_in[2];
  const unsigned char* mask = (const unsigned char*)d_in[3];
  const float* pair_bias = (const float*)d_in[4];
  const float* w_f1 = (const float*)d_in[5];  const float* b_f1 = (const float*)d_in[6];
  const float* w_f2 = (const float*)d_in[7];  const float* b_f2 = (const float*)d_in[8];
  const float* cn1_ws = (const float*)d_in[9];  const float* cn1_bs = (const float*)d_in[10];
  const float* cn1_wb = (const float*)d_in[11]; const float* cn1_bb = (const float*)d_in[12];
  const float* wq = (const float*)d_in[13]; const float* wk = (const float*)d_in[14];
  const float* wv = (const float*)d_in[15];
  const float* wqp = (const float*)d_in[16]; const float* wkp = (const float*)d_in[17];
  const float* wvp = (const float*)d_in[18];
  const float* gamma = (const float*)d_in[19];
  const float* wo = (const float*)d_in[20]; const float* bo = (const float*)d_in[21];
  const float* cn2_ws = (const float*)d_in[22]; const float* cn2_bs = (const float*)d_in[23];
  const float* cn2_wb = (const float*)d_in[24]; const float* cn2_bb = (const float*)d_in[25];
  const float* wg1 = (const float*)d_in[26]; const float* bg1 = (const float*)d_in[27];
  const float* wg2 = (const float*)d_in[28]; const float* bg2 = (const float*)d_in[29];
  const float* wg3 = (const float*)d_in[30]; const float* bg3 = (const float*)d_in[31];
  const float* cn3_ws = (const float*)d_in[32]; const float* cn3_bs = (const float*)d_in[33];
  const float* cn3_wb = (const float*)d_in[34]; const float* cn3_bb = (const float*)d_in[35];
  const float* w_vel = (const float*)d_in[36];

  const int Nn = NTOK, Dd = DIM;
  const size_t ND = (size_t)Nn * Dd, NP = (size_t)Nn * 384, HNsz = (size_t)NH * Nn;

  float* ws = (float*)d_ws;
  size_t off = 0;
  auto alloc = [&](size_t e) { size_t o = off; off += (e + 3) & ~(size_t)3; return ws + o; };
  auto allocb = [&](size_t e) { return (bf16*)alloc((e + 1) / 2); };

  // f32 buffers
  float* local0 = alloc(ND);
  float* s1b = alloc(ND);
  float* s2b = alloc(ND);
  float* qb = alloc(ND); float* kb = alloc(ND); float* vb = alloc(ND);
  float* qp = alloc(NP); float* kp = alloc(NP); float* vp = alloc(NP);
  float* Rb = alloc((size_t)Nn * 9);
  float* tb = alloc((size_t)Nn * 3);
  float* anb = alloc(HNsz);
  float* bnb = alloc(HNsz);
  float* attnb = alloc(HNsz * KHAT);
  // bf16 activations
  bf16* embB  = allocb((size_t)Nn * FD);
  bf16* h1B   = allocb((size_t)Nn * D2);   // also g1s
  bf16* condB = allocb(ND);
  bf16* aiB   = allocb(ND);                // also x, ln
  bf16* bigB  = allocb((size_t)Nn * D2);   // updin, then prod
  bf16* g2B   = allocb((size_t)Nn * D2);
  bf16* Ahat  = allocb(HNsz * KHAT);
  bf16* Bhat  = allocb(HNsz * KHAT);
  bf16* Vhat  = allocb(HNsz * KHAT);

  // pre-transposed bf16 weights
  auto prep = [&](const float* w, int K, int N) {
    bf16* wt = allocb((size_t)K * N);
    tconv_kernel<<<dim3((N + 31) / 32, (K + 31) / 32), dim3(32, 8), 0, stream>>>(
        w, wt, K, N);
    return wt;
  };
  bf16* w_f1T = prep(w_f1, FD, D2);
  bf16* w_f2T = prep(w_f2, D2, Dd);
  bf16* cn1_wsT = prep(cn1_ws, Dd, Dd); bf16* cn1_wbT = prep(cn1_wb, Dd, Dd);
  bf16* cn2_wsT = prep(cn2_ws, Dd, Dd); bf16* cn2_wbT = prep(cn2_wb, Dd, Dd);
  bf16* cn3_wsT = prep(cn3_ws, Dd, Dd); bf16* cn3_wbT = prep(cn3_wb, Dd, Dd);
  bf16* wqT = prep(wq, Dd, Dd); bf16* wkT = prep(wk, Dd, Dd);
  bf16* wvT = prep(wv, Dd, Dd);
  bf16* wqpT = prep(wqp, Dd, 384); bf16* wkpT = prep(wkp, Dd, 384);
  bf16* wvpT = prep(wvp, Dd, 384);
  bf16* woT = prep(wo, 1536, Dd);
  bf16* wg1T = prep(wg1, Dd, D2); bf16* wg2T = prep(wg2, Dd, D2);
  bf16* wg3T = prep(wg3, D2, Dd);
  bf16* w_velT = prep(w_vel, Dd, 42);

  cvt_kernel<<<((int)ND + 255) / 256, 256, 0, stream>>>(cond, condB, (int)ND);

  float* out_local = (float*)d_out;
  float* out_vel = out_local + ND;

  auto gemm = [&](const bf16* A, const bf16* Bt, const float* bias,
                  const float* res, void* C, int M, int N, int K, int act,
                  int outbf) {
    dim3 grid((N + GTN - 1) / GTN, (M + GTM - 1) / GTM);
    gemm_kernel<<<grid, 256, 0, stream>>>(A, Bt, bias, res, C, M, N, K, act, outbf);
  };

  // 1) frequency embedding (bf16)
  embed_kernel<<<(Nn * 42 * 64 + 255) / 256, 256, 0, stream>>>(pos, embB);
  // 2) local += silu(emb@w_f1+b_f1)@w_f2 + b_f2
  gemm(embB, w_f1T, b_f1, nullptr, h1B, Nn, D2, FD, 1, 1);
  gemm(h1B, w_f2T, b_f2, in_local, local0, Nn, Dd, D2, 0, 0);
  // 3) ai = cond_norm(local0)
  gemm(condB, cn1_wsT, cn1_bs, nullptr, s1b, Nn, Dd, Dd, 2, 0);
  gemm(condB, cn1_wbT, cn1_bb, nullptr, s2b, Nn, Dd, Dd, 0, 0);
  ln_mod_kernel<<<Nn, 256, 0, stream>>>(local0, s1b, s2b, aiB, Dd);
  // 4) q/k/v + point projections (f32 out)
  gemm(aiB, wqT, nullptr, nullptr, qb, Nn, Dd, Dd, 0, 0);
  gemm(aiB, wkT, nullptr, nullptr, kb, Nn, Dd, Dd, 0, 0);
  gemm(aiB, wvT, nullptr, nullptr, vb, Nn, Dd, Dd, 0, 0);
  gemm(aiB, wqpT, nullptr, nullptr, qp, Nn, 384, Dd, 0, 0);
  gemm(aiB, wkpT, nullptr, nullptr, kp, Nn, 384, Dd, 0, 0);
  gemm(aiB, wvpT, nullptr, nullptr, vp, Nn, 384, Dd, 0, 0);
  // 5) frames + local->global points (in place)
  frames_kernel<<<(Nn + 255) / 256, 256, 0, stream>>>(pos, Rb, tb);
  toglob_kernel<<<(Nn * NH * NPTS + 255) / 256, 256, 0, stream>>>(qp, Rb, tb);
  toglob_kernel<<<(Nn * NH * NPTS + 255) / 256, 256, 0, stream>>>(kp, Rb, tb);
  toglob_kernel<<<(Nn * NH * NPTS + 255) / 256, 256, 0, stream>>>(vp, Rb, tb);
  // 6) pack augmented q/k/v + point norms
  pack_qkv_kernel<<<(Nn * NH + 255) / 256, 256, 0, stream>>>(
      qb, kb, vb, qp, kp, vp, gamma, Ahat, Bhat, Vhat, anb, bnb);
  // 7) fused IPA attention
  attn_kernel<<<dim3(Nn / 128, NH), 256, 0, stream>>>(
      Ahat, Bhat, Vhat, anb, bnb, pair_bias, mask, attnb, Nn);
  // 8) back-transform + concat [o, opl, onorm] (bf16)
  pack_upd_kernel<<<(Nn * NH + 255) / 256, 256, 0, stream>>>(attnb, Rb, tb, bigB);
  // 9) local += upd @ wo + bo
  gemm(bigB, woT, bo, local0, local0, Nn, Dd, 1536, 0, 0);
  // 10) x = cond_norm(local)
  gemm(condB, cn2_wsT, cn2_bs, nullptr, s1b, Nn, Dd, Dd, 2, 0);
  gemm(condB, cn2_wbT, cn2_bb, nullptr, s2b, Nn, Dd, Dd, 0, 0);
  ln_mod_kernel<<<Nn, 256, 0, stream>>>(local0, s1b, s2b, aiB, Dd);
  // 11) gated MLP: local += (silu(x@wg1+bg1) * (x@wg2+bg2)) @ wg3 + bg3
  gemm(aiB, wg1T, bg1, nullptr, h1B, Nn, D2, Dd, 1, 1);
  gemm(aiB, wg2T, bg2, nullptr, g2B, Nn, D2, Dd, 0, 1);
  mul_kernel<<<((int)(Nn * D2) + 255) / 256, 256, 0, stream>>>(
      h1B, g2B, bigB, Nn * D2);
  gemm(bigB, wg3T, bg3, local0, out_local, Nn, Dd, D2, 0, 0);
  // 12) velocity = cond_norm(local) @ w_vel
  gemm(condB, cn3_wsT, cn3_bs, nullptr, s1b, Nn, Dd, Dd, 2, 0);
  gemm(condB, cn3_wbT, cn3_bb, nullptr, s2b, Nn, Dd, Dd, 0, 0);
  ln_mod_kernel<<<Nn, 256, 0, stream>>>(out_local, s1b, s2b, aiB, Dd);
  gemm(aiB, w_velT, nullptr, nullptr, out_vel, Nn, 42, Dd, 0, 0);
}
